// TransformerBlock_10462540333625
// MI455X (gfx1250) — matrix operand order repaired and checked by reading, not compiled, not run on hardware
//
#include <hip/hip_runtime.h>

// ---------------------------------------------------------------------------
// Transformer block for MI455X (gfx1250, wave32, WMMA).
// Pipeline: convert/transpose weights to bf16 -> LN1 -> QKV WMMA GEMMs
//           -> causal flash attention (WMMA scores + WMMA P@V via LDS)
//           -> LN2 -> FFN1 (bias+relu) -> FFN2 (bias+residual) -> d_out (f32)
// GEMM: 8-wave workgroup, B-tile staged in LDS via GLOBAL_LOAD_ASYNC_TO_LDS
// (double buffered, s_wait_asynccnt), wave computes 32Mx64N with 8 WMMA/K-step.
// ---------------------------------------------------------------------------

#define DEV __device__ __forceinline__

typedef __attribute__((ext_vector_type(16))) __bf16 v16bf;
typedef __attribute__((ext_vector_type(8)))  __bf16 v8bf;
typedef __attribute__((ext_vector_type(8)))  float  v8f;

union ABfrag { v16bf v; v8bf h[2]; };

constexpr int Bb   = 4;
constexpr int Ss   = 2048;
constexpr int Dd   = 1024;
constexpr int Ff   = 4096;
constexpr int Mtok = Bb * Ss;   // 8192 token rows

#if defined(__has_builtin)
#if __has_builtin(__builtin_amdgcn_global_load_async_to_lds_b128) && \
    __has_builtin(__builtin_amdgcn_s_wait_asynccnt)
#define USE_ASYNC_LDS 1
#endif
#endif

#ifdef USE_ASYNC_LDS
typedef int v4i_g __attribute__((vector_size(16)));
typedef __attribute__((address_space(1))) v4i_g* as1_v4i;   // global int4*
typedef __attribute__((address_space(3))) v4i_g* as3_v4i;   // LDS int4*
#endif

// round-to-nearest-even f32 -> bf16 bits
DEV unsigned short f2bf(float f) {
  unsigned int u = __float_as_uint(f);
  u += 0x7fffu + ((u >> 16) & 1u);
  return (unsigned short)(u >> 16);
}

DEV v8f wmma_bf16(const ABfrag& a, const ABfrag& b, v8f c) {
  return __builtin_amdgcn_wmma_f32_16x16x32_bf16(
      /*neg_a=*/false, a.v, /*neg_b=*/false, b.v,
      /*c_mod=*/(short)0, c, /*reuse_a=*/false, /*reuse_b=*/false);
}

DEV v8f vzero8() { v8f z = {0.f,0.f,0.f,0.f,0.f,0.f,0.f,0.f}; return z; }

// ---------------------------------------------------------------------------
// Weight convert + transpose: w[K][N] f32 -> wt[N][K] bf16
// ---------------------------------------------------------------------------
__global__ void __launch_bounds__(256)
transpose_convert_kernel(const float* __restrict__ w,
                         unsigned short* __restrict__ wt,
                         int K, int N) {
  int k = blockIdx.y * 256 + threadIdx.x;
  int n = blockIdx.x;
  if (k < K) wt[(size_t)n * K + k] = f2bf(w[(size_t)k * N + n]);
}

// ---------------------------------------------------------------------------
// LayerNorm: one wave per row, D = 1024, output bf16
// ---------------------------------------------------------------------------
__global__ void __launch_bounds__(32)
layernorm_bf16_kernel(const float* __restrict__ x,
                      const float* __restrict__ gamma,
                      const float* __restrict__ beta,
                      unsigned short* __restrict__ out) {
  const int row  = blockIdx.x;
  const int lane = threadIdx.x & 31;
  const float* xr = x + (size_t)row * Dd;
  float s = 0.f, ss = 0.f;
  #pragma unroll
  for (int i = lane; i < Dd; i += 32) {
    float v = xr[i];
    s += v; ss += v * v;
  }
  #pragma unroll
  for (int o = 16; o > 0; o >>= 1) {
    s  += __shfl_xor(s,  o, 32);
    ss += __shfl_xor(ss, o, 32);
  }
  const float mean = s * (1.f / Dd);
  const float var  = ss * (1.f / Dd) - mean * mean;
  const float inv  = rsqrtf(var + 1e-5f);
  #pragma unroll
  for (int i = lane; i < Dd; i += 32) {
    float v = (xr[i] - mean) * inv * gamma[i] + beta[i];
    out[(size_t)row * Dd + i] = f2bf(v);
  }
}

// ---------------------------------------------------------------------------
// B-tile staging helpers (64 N-rows x 32 K bf16 = 4 KB per buffer).
// Thread t loads 16B: row r = t>>2, chunk cq = t&3.
// ---------------------------------------------------------------------------
DEV void stage_B(const unsigned short* __restrict__ Bt,
                 unsigned short* lds, int n0, int K, int kk, int tid) {
  const int r  = tid >> 2;
  const int cq = tid & 3;
  const unsigned short* src = Bt + (size_t)(n0 + r) * K + kk + cq * 8;
  unsigned short* dst = lds + r * 32 + cq * 8;
#ifdef USE_ASYNC_LDS
  __builtin_amdgcn_global_load_async_to_lds_b128(
      (as1_v4i)src, (as3_v4i)dst, 0, 0);
#else
  *(v8bf*)dst = *(const v8bf*)src;
#endif
}

DEV void stage_wait() {
#ifdef USE_ASYNC_LDS
  __builtin_amdgcn_s_wait_asynccnt(0);
#endif
  __syncthreads();
}

// ---------------------------------------------------------------------------
// WMMA GEMM: C[M,N] = A[M,K](bf16, row-major) x Bt[N,K](bf16, N-major)
// Block: 256 threads (8 waves) -> 256M x 64N. Wave: 32M x 64N (8 accums).
// B tile shared via LDS, double buffered.
// MODE 0: f32 out (+bias, +resid)    MODE 1: bf16 row-major out (+bias,+relu)
// MODE 2: bf16 transposed out (Vt[b][n][s]), N must be Dd
// ---------------------------------------------------------------------------
template<int MODE, int RELU, int HASBIAS, int HASRESID>
__global__ void __launch_bounds__(256)
gemm_bf16_kernel(const unsigned short* __restrict__ A,
                 const unsigned short* __restrict__ Bt,
                 const float* __restrict__ bias,
                 const float* __restrict__ resid,
                 float* __restrict__ outF,
                 unsigned short* __restrict__ outB,
                 int N, int K) {
  __shared__ __attribute__((aligned(16))) unsigned short Btile[2][64 * 32];

  const int tid  = threadIdx.x;
  const int wave = tid >> 5;
  const int lane = tid & 31;
  const int hi   = lane >> 4;
  const int ln   = lane & 15;
  const int m0   = blockIdx.y * 256 + wave * 32;
  const int n0   = blockIdx.x << 6;

  v8f c0[4], c1[4];
  #pragma unroll
  for (int t = 0; t < 4; ++t) { c0[t] = vzero8(); c1[t] = vzero8(); }

  const unsigned short* Arow0 = A + (size_t)(m0 + ln) * K;
  const unsigned short* Arow1 = A + (size_t)(m0 + 16 + ln) * K;

  stage_B(Bt, &Btile[0][0], n0, K, 0, tid);
  stage_wait();

  for (int kk = 0; kk < K; kk += 32) {
    const int cur = (kk >> 5) & 1;
    if (kk + 32 < K) stage_B(Bt, &Btile[cur ^ 1][0], n0, K, kk + 32, tid);

    ABfrag a0, a1;
    a0.h[0] = *(const v8bf*)(Arow0 + kk + hi * 8);
    a0.h[1] = *(const v8bf*)(Arow0 + kk + 16 + hi * 8);
    a1.h[0] = *(const v8bf*)(Arow1 + kk + hi * 8);
    a1.h[1] = *(const v8bf*)(Arow1 + kk + 16 + hi * 8);
    if (kk + 64 < K) {
      __builtin_prefetch(Arow0 + kk + 64, 0, 1);
      __builtin_prefetch(Arow1 + kk + 64, 0, 1);
    }

    // Load all 4 B fragments first so one wait covers all DS reads,
    // then issue the 8 WMMAs back-to-back.
    const unsigned short* lb = &Btile[cur][0];
    ABfrag b[4];
    #pragma unroll
    for (int tn = 0; tn < 4; ++tn) {
      const unsigned short* p = lb + (tn * 16 + ln) * 32 + hi * 8;
      b[tn].h[0] = *(const v8bf*)(p);
      b[tn].h[1] = *(const v8bf*)(p + 16);
    }
    #pragma unroll
    for (int tn = 0; tn < 4; ++tn) {
      c0[tn] = wmma_bf16(a0, b[tn], c0[tn]);
      c1[tn] = wmma_bf16(a1, b[tn], c1[tn]);
    }

    if (kk + 32 < K) stage_wait();
  }

  #pragma unroll
  for (int half = 0; half < 2; ++half) {
    const v8f* cc = half ? c1 : c0;
    const int mb = m0 + half * 16;
    #pragma unroll
    for (int t = 0; t < 4; ++t) {
      const int n = n0 + t * 16 + ln;
      float bv = 0.f;
      if constexpr (HASBIAS) bv = bias[n];
      #pragma unroll
      for (int v = 0; v < 8; ++v) {
        const int m = mb + hi * 8 + v;
        float val = cc[t][v] + bv;
        if constexpr (RELU) val = fmaxf(val, 0.f);
        if constexpr (MODE == 0) {
          if constexpr (HASRESID) val += resid[(size_t)m * N + n];
          outF[(size_t)m * N + n] = val;
        } else if constexpr (MODE == 1) {
          outB[(size_t)m * N + n] = f2bf(val);
        } else {
          const int bidx = m >> 11;          // m / Ss
          const int s    = m & (Ss - 1);
          outB[((size_t)bidx * Dd + n) * Ss + s] = f2bf(val);
        }
      }
    }
  }
}

// ---------------------------------------------------------------------------
// Attention helpers
// ---------------------------------------------------------------------------
DEV void score32(const unsigned short* __restrict__ Qrow,
                 const unsigned short* __restrict__ K0,
                 const unsigned short* __restrict__ K1,
                 int hi, v8f& s0, v8f& s1) {
  s0 = vzero8();
  s1 = vzero8();
  for (int kk = 0; kk < Dd; kk += 32) {
    ABfrag a, b0, b1;
    a.h[0] = *(const v8bf*)(Qrow + kk + hi * 8);
    a.h[1] = *(const v8bf*)(Qrow + kk + 16 + hi * 8);
    const unsigned short* p0 = K0 + kk + hi * 8;
    const unsigned short* p1 = K1 + kk + hi * 8;
    b0.h[0] = *(const v8bf*)(p0);  b0.h[1] = *(const v8bf*)(p0 + 16);
    b1.h[0] = *(const v8bf*)(p1);  b1.h[1] = *(const v8bf*)(p1 + 16);
    s0 = wmma_bf16(a, b0, s0);
    s1 = wmma_bf16(a, b1, s1);
  }
}

// ---------------------------------------------------------------------------
// Causal attention: one wave per (batch, 16-row q block).
// Pass1: online row max/sum. Pass2: normalized P -> LDS (16 x Ss bf16).
// Pass3: O = P @ V via WMMA (Vt is [b][d][s]), write x1 = x + O.
// ---------------------------------------------------------------------------
__global__ void __launch_bounds__(32)
attn_kernel(const unsigned short* __restrict__ Q,
            const unsigned short* __restrict__ Km,
            const unsigned short* __restrict__ Vt,
            const float* __restrict__ x,
            float* __restrict__ x1) {
  extern __shared__ unsigned short P[];   // [16][Ss] bf16 probabilities

  const int lane = threadIdx.x & 31;
  const int hi   = lane >> 4;
  const int ln   = lane & 15;
  const int nqb  = Ss / 16;
  const int bidx = blockIdx.x / nqb;
  const int qb   = blockIdx.x % nqb;
  const int q0   = qb * 16;               // local (in-batch) first q row
  const size_t rowbase = (size_t)bidx * Ss;
  const unsigned short* Qrow  = Q + (rowbase + q0 + ln) * Dd;
  const unsigned short* Kbase = Km + rowbase * Dd;
  const float scale = 0.03125f;           // 1/sqrt(1024)
  const int nkb = (q0 + 16 + 31) / 32;    // causal key blocks of 32

  float rm[8], rs[8];
  #pragma unroll
  for (int v = 0; v < 8; ++v) { rm[v] = -__builtin_inff(); rs[v] = 0.f; }

  // ---- Pass 1: statistics ----
  for (int ib = 0; ib < nkb; ++ib) {
    const int kb = ib * 32;
    v8f s0, s1;
    score32(Qrow, Kbase + (size_t)(kb + ln) * Dd,
                  Kbase + (size_t)(kb + 16 + ln) * Dd, hi, s0, s1);
    const int k0i = kb + ln, k1i = kb + 16 + ln;
    #pragma unroll
    for (int v = 0; v < 8; ++v) {
      const int q = q0 + hi * 8 + v;
      const float e0 = (k0i <= q) ? s0[v] * scale : -__builtin_inff();
      const float e1 = (k1i <= q) ? s1[v] * scale : -__builtin_inff();
      float bm = fmaxf(e0, e1);
      #pragma unroll
      for (int o = 1; o < 16; o <<= 1) bm = fmaxf(bm, __shfl_xor(bm, o, 16));
      const float nm = fmaxf(rm[v], bm);
      float p = __expf(e0 - nm) + __expf(e1 - nm);
      #pragma unroll
      for (int o = 1; o < 16; o <<= 1) p += __shfl_xor(p, o, 16);
      rs[v] = rs[v] * __expf(rm[v] - nm) + p;
      rm[v] = nm;
    }
  }

  float inv[8];
  #pragma unroll
  for (int v = 0; v < 8; ++v) inv[v] = 1.f / rs[v];

  // ---- Pass 2: normalized probabilities into LDS ----
  for (int ib = 0; ib < nkb; ++ib) {
    const int kb = ib * 32;
    v8f s0, s1;
    score32(Qrow, Kbase + (size_t)(kb + ln) * Dd,
                  Kbase + (size_t)(kb + 16 + ln) * Dd, hi, s0, s1);
    const int k0i = kb + ln, k1i = kb + 16 + ln;
    #pragma unroll
    for (int v = 0; v < 8; ++v) {
      const int q   = q0 + hi * 8 + v;
      const int row = hi * 8 + v;
      const float p0 = (k0i <= q) ? __expf(s0[v] * scale - rm[v]) * inv[v] : 0.f;
      const float p1 = (k1i <= q) ? __expf(s1[v] * scale - rm[v]) * inv[v] : 0.f;
      P[(size_t)row * Ss + k0i] = f2bf(p0);
      P[(size_t)row * Ss + k1i] = f2bf(p1);
    }
  }
  __syncthreads();

  // ---- Pass 3: O = P @ V, residual add, write x1 ----
  for (int v0 = 0; v0 < Dd; v0 += 64) {
    v8f c[4];
    #pragma unroll
    for (int t = 0; t < 4; ++t) c[t] = vzero8();
    for (int ib = 0; ib < nkb; ++ib) {
      const int kb = ib * 32;
      ABfrag a;
      a.h[0] = *(const v8bf*)&P[(size_t)ln * Ss + kb + hi * 8];
      a.h[1] = *(const v8bf*)&P[(size_t)ln * Ss + kb + 16 + hi * 8];
      ABfrag b[4];
      #pragma unroll
      for (int t = 0; t < 4; ++t) {
        const unsigned short* Vr =
            Vt + ((size_t)bidx * Dd + v0 + t * 16 + ln) * Ss + kb + hi * 8;
        b[t].h[0] = *(const v8bf*)(Vr);
        b[t].h[1] = *(const v8bf*)(Vr + 16);
      }
      #pragma unroll
      for (int t = 0; t < 4; ++t) c[t] = wmma_bf16(a, b[t], c[t]);
    }
    #pragma unroll
    for (int t = 0; t < 4; ++t) {
      const int n = v0 + t * 16 + ln;
      #pragma unroll
      for (int v = 0; v < 8; ++v) {
        const size_t m = rowbase + q0 + hi * 8 + v;
        x1[m * Dd + n] = x[m * Dd + n] + c[t][v];
      }
    }
  }
}

// ---------------------------------------------------------------------------
// Host-side orchestration
// ---------------------------------------------------------------------------
extern "C" void kernel_launch(void* const* d_in, const int* in_sizes, int n_in,
                              void* d_out, int out_size, void* d_ws, size_t ws_size,
                              hipStream_t stream) {
  (void)in_sizes; (void)n_in; (void)out_size; (void)ws_size;

  const float* x   = (const float*)d_in[0];
  const float* Qw  = (const float*)d_in[1];
  const float* Kw  = (const float*)d_in[2];
  const float* Vw  = (const float*)d_in[3];
  const float* w1  = (const float*)d_in[4];
  const float* b1  = (const float*)d_in[5];
  const float* w2  = (const float*)d_in[6];
  const float* b2  = (const float*)d_in[7];
  const float* g1  = (const float*)d_in[8];
  const float* be1 = (const float*)d_in[9];
  const float* g2  = (const float*)d_in[10];
  const float* be2 = (const float*)d_in[11];
  float* out = (float*)d_out;

  char* ws = (char*)d_ws;
  size_t off = 0;
  auto alloc = [&](size_t bytes) -> void* {
    void* p = ws + off;
    off += (bytes + 255) & ~(size_t)255;
    return p;
  };

  unsigned short* Qwt = (unsigned short*)alloc((size_t)Dd * Dd * 2);
  unsigned short* Kwt = (unsigned short*)alloc((size_t)Dd * Dd * 2);
  unsigned short* Vwt = (unsigned short*)alloc((size_t)Dd * Dd * 2);
  unsigned short* w1t = (unsigned short*)alloc((size_t)Ff * Dd * 2);
  unsigned short* w2t = (unsigned short*)alloc((size_t)Dd * Ff * 2);
  unsigned short* xn  = (unsigned short*)alloc((size_t)Mtok * Dd * 2); // ln1 out, reused for ln2
  unsigned short* Qb  = (unsigned short*)alloc((size_t)Mtok * Dd * 2);
  unsigned short* Kb  = (unsigned short*)alloc((size_t)Mtok * Dd * 2);
  unsigned short* Vtb = (unsigned short*)alloc((size_t)Mtok * Dd * 2); // [B][Dd][Ss]
  float*          x1  = (float*)alloc((size_t)Mtok * Dd * 4);
  unsigned short* act = (unsigned short*)alloc((size_t)Mtok * Ff * 2);

  // 1) weights -> bf16 transposed [N][K]
  transpose_convert_kernel<<<dim3(Dd, Dd / 256), 256, 0, stream>>>(Qw, Qwt, Dd, Dd);
  transpose_convert_kernel<<<dim3(Dd, Dd / 256), 256, 0, stream>>>(Kw, Kwt, Dd, Dd);
  transpose_convert_kernel<<<dim3(Dd, Dd / 256), 256, 0, stream>>>(Vw, Vwt, Dd, Dd);
  transpose_convert_kernel<<<dim3(Ff, Dd / 256), 256, 0, stream>>>(w1, w1t, Dd, Ff);
  transpose_convert_kernel<<<dim3(Dd, Ff / 256), 256, 0, stream>>>(w2, w2t, Ff, Dd);

  // 2) LN1
  layernorm_bf16_kernel<<<Mtok, 32, 0, stream>>>(x, g1, be1, xn);

  // 3) Q, K (bf16 row-major), V (bf16 transposed)
  gemm_bf16_kernel<1,0,0,0><<<dim3(Dd / 64, Mtok / 256), 256, 0, stream>>>(
      xn, Qwt, nullptr, nullptr, nullptr, Qb, Dd, Dd);
  gemm_bf16_kernel<1,0,0,0><<<dim3(Dd / 64, Mtok / 256), 256, 0, stream>>>(
      xn, Kwt, nullptr, nullptr, nullptr, Kb, Dd, Dd);
  gemm_bf16_kernel<2,0,0,0><<<dim3(Dd / 64, Mtok / 256), 256, 0, stream>>>(
      xn, Vwt, nullptr, nullptr, nullptr, Vtb, Dd, Dd);

  // 4) causal attention + residual -> x1
  attn_kernel<<<Bb * (Ss / 16), 32, (size_t)16 * Ss * 2, stream>>>(
      Qb, Kb, Vtb, x, x1);

  // 5) LN2 (reuse xn)
  layernorm_bf16_kernel<<<Mtok, 32, 0, stream>>>(x1, g2, be2, xn);

  // 6) FFN1: relu(h @ w1 + b1) -> act (bf16)
  gemm_bf16_kernel<1,1,1,0><<<dim3(Ff / 64, Mtok / 256), 256, 0, stream>>>(
      xn, w1t, b1, nullptr, nullptr, act, Ff, Dd);

  // 7) FFN2: act @ w2 + b2 + x1 -> d_out (f32)
  gemm_bf16_kernel<0,0,1,1><<<dim3(Dd / 64, Mtok / 256), 256, 0, stream>>>(
      act, w2t, b2, x1, out, nullptr, Dd, Ff);
}
